// BistableRecurrentCellLayer_57913339019733
// MI455X (gfx1250) — compile-verified, matrix operand order checked by reading, not run on hardware
//
#include <hip/hip_runtime.h>
#include <hip/hip_bf16.h>
#include <math.h>

// BRC layer: B=64, T=512, D_IN=512, H=512
// Phase 1: triple GEMM  xz/xr/xh = x @ {kz,kr,kh}   (WMMA bf16 -> f32 acc)
// Phase 2: sequential elementwise scan over T (pure HBM streaming)

#define BB 64
#define TT 512
#define KK 512   // D_IN
#define HH 512
#define MM (BB * TT)   // 32768 rows of x flattened

typedef __attribute__((ext_vector_type(16))) __bf16 v16bf;
typedef __attribute__((ext_vector_type(8)))  float  v8f;

#define MT 4  // 16-row M tiles per wave -> wave owns 64 x 16 output per matrix

// ---------------------------------------------------------------------------
// Phase 1: each wave computes a 64x16 tile of xz, xr, xh.
// A fragment (16x32 bf16) per the CDNA5 layout:
//   lanes 0-15 : M = lane,    K elems {0..7, 16..23}  (+kk)
//   lanes 16-31: M = lane-16, K elems {8..15, 24..31} (+kk)
// element e of the v16bf maps to k = (e<8 ? e : e+8) + 8*half.
// B fragment (32x16, B stored row-major [K,H]) mirrors with N = lane&15.
// ---------------------------------------------------------------------------
__global__ __launch_bounds__(256)
void brc_gemm3_wmma(const float* __restrict__ x,    // [MM, KK]
                    const float* __restrict__ wz,   // [KK, HH]
                    const float* __restrict__ wr,   // [KK, HH]
                    const float* __restrict__ wh,   // [KK, HH]
                    float* __restrict__ oz,         // [MM, HH]
                    float* __restrict__ orr,        // [MM, HH]
                    float* __restrict__ oh)         // [MM, HH]
{
    const int wave = blockIdx.x * (blockDim.x >> 5) + (threadIdx.x >> 5);
    const int nt   = wave & 31;        // 32 tiles cover H=512
    const int mt   = wave >> 5;        // 512 supertiles cover M=32768 (64 rows each)
    const int lane = threadIdx.x & 31;
    const int half = lane >> 4;        // 0 or 1
    const int l    = lane & 15;
    const int n    = nt * 16 + l;      // output column
    const int m0   = mt * 64;

    v8f accz[MT] = {};
    v8f accr[MT] = {};
    v8f acch[MT] = {};

    for (int kk = 0; kk < KK; kk += 32) {
        const int kb = kk + half * 8;

        // --- B fragments for the three weight matrices (shared across MT) ---
        v16bf bz_f, br_f, bh_f;
#pragma unroll
        for (int e = 0; e < 16; ++e) {
            const int k = kb + (e < 8 ? e : e + 8);
            bz_f[e] = (__bf16)wz[(size_t)k * HH + n];
            br_f[e] = (__bf16)wr[(size_t)k * HH + n];
            bh_f[e] = (__bf16)wh[(size_t)k * HH + n];
        }

#pragma unroll
        for (int mi = 0; mi < MT; ++mi) {
            const int row = m0 + mi * 16 + l;
            v16bf a_f;
#pragma unroll
            for (int e = 0; e < 16; ++e) {
                const int k = kb + (e < 8 ? e : e + 8);
                a_f[e] = (__bf16)x[(size_t)row * KK + k];
            }
            // D = A*B + C   (neg_a, A, neg_b, B, c_mod, C, reuse_a, reuse_b)
            accz[mi] = __builtin_amdgcn_wmma_f32_16x16x32_bf16(
                false, a_f, false, bz_f, (short)0, accz[mi], false, false);
            accr[mi] = __builtin_amdgcn_wmma_f32_16x16x32_bf16(
                false, a_f, false, br_f, (short)0, accr[mi], false, false);
            acch[mi] = __builtin_amdgcn_wmma_f32_16x16x32_bf16(
                false, a_f, false, bh_f, (short)0, acch[mi], false, false);
        }
    }

    // --- store C/D fragments: VGPR j -> row (j + 8*half), lane -> column ---
#pragma unroll
    for (int mi = 0; mi < MT; ++mi) {
#pragma unroll
        for (int j = 0; j < 8; ++j) {
            const int row = m0 + mi * 16 + half * 8 + j;
            const size_t idx = (size_t)row * HH + n;
            oz [idx] = accz[mi][j];
            orr[idx] = accr[mi][j];
            oh [idx] = acch[mi][j];
        }
    }
}

// ---------------------------------------------------------------------------
// Phase 2: sequential scan over T. Each thread owns TWO adjacent h columns
// (float2 traffic, two independent dependency chains). xh lives in `out` and
// is overwritten in place with the hidden state.
// ---------------------------------------------------------------------------
__global__ __launch_bounds__(256)
void brc_scan(const float* __restrict__ xz,
              const float* __restrict__ xr,
              float* __restrict__ out,              // holds xh, overwritten with h
              const float* __restrict__ h0,
              const float* __restrict__ mzv,
              const float* __restrict__ mrv,
              const float* __restrict__ bzv,
              const float* __restrict__ brv)
{
    const int tid  = blockIdx.x * blockDim.x + threadIdx.x;  // 0 .. B*H/2-1
    const int b    = tid >> 8;                               // H/2 = 256 pairs per batch
    const int hcol = (tid & 255) * 2;

    const float2 mz = *(const float2*)(mzv + hcol);
    const float2 mr = *(const float2*)(mrv + hcol);
    const float2 bz = *(const float2*)(bzv + hcol);
    const float2 br = *(const float2*)(brv + hcol);
    float2 h = *(const float2*)(h0 + (size_t)b * HH + hcol);

    const size_t base = (size_t)b * TT * HH + hcol;
    for (int t = 0; t < TT; ++t) {
        const size_t off = base + (size_t)t * HH;
        const float2 vz = *(const float2*)(xz + off);
        const float2 vr = *(const float2*)(xr + off);
        const float2 vh = *(const float2*)(out + off);

        float r0 = tanhf(vr.x + br.x + h.x * mr.x) + 1.0f;
        float r1 = tanhf(vr.y + br.y + h.y * mr.y) + 1.0f;
        float z0 = 1.0f / (1.0f + __expf(-(vz.x + bz.x + h.x * mz.x)));
        float z1 = 1.0f / (1.0f + __expf(-(vz.y + bz.y + h.y * mz.y)));
        float n0 = tanhf(vh.x + r0 * h.x);
        float n1 = tanhf(vh.y + r1 * h.y);
        h.x = z0 * h.x + (1.0f - z0) * n0;
        h.y = z1 * h.y + (1.0f - z1) * n1;

        *(float2*)(out + off) = h;
    }
}

extern "C" void kernel_launch(void* const* d_in, const int* in_sizes, int n_in,
                              void* d_out, int out_size, void* d_ws, size_t ws_size,
                              hipStream_t stream) {
    const float* x  = (const float*)d_in[0];
    const float* h0 = (const float*)d_in[1];
    const float* kz = (const float*)d_in[2];
    const float* kr = (const float*)d_in[3];
    const float* kh = (const float*)d_in[4];
    const float* mz = (const float*)d_in[5];
    const float* mr = (const float*)d_in[6];
    const float* bz = (const float*)d_in[7];
    const float* br = (const float*)d_in[8];

    float* out = (float*)d_out;                    // [B,T,H] -> xh then h
    float* xz  = (float*)d_ws;                     // [M,H]  64 MB
    float* xr  = xz + (size_t)MM * HH;             // [M,H]  64 MB

    // Phase 1: (M/64)*(H/16) = 512*32 = 16384 waves, 8 waves (256 thr) / block
    brc_gemm3_wmma<<<16384 / 8, 256, 0, stream>>>(x, kz, kr, kh, xz, xr, out);

    // Phase 2: B*H/2 = 16384 threads
    brc_scan<<<(BB * HH / 2) / 256, 256, 0, stream>>>(xz, xr, out, h0, mz, mr, bz, br);
}
